// Seq2SeqRNNAttn_25589415150271
// MI455X (gfx1250) — compile-verified, matrix operand order checked by reading, not run on hardware
//
#include <hip/hip_runtime.h>
#include <hip/hip_bf16.h>

// ---------------- constants ----------------
constexpr int kB  = 512;   // batch
constexpr int kTE = 56;    // encoder time steps
constexpr int kTD = 28;    // decoder time steps
constexpr int kNH = 512;   // encoder hidden
constexpr int kD  = 576;   // decoder hidden
constexpr int kKX = 32;    // padded encoder input feature dim (20 -> 32)
constexpr int kKA = 544;   // padded attn_out dim (513 -> 544)

typedef __attribute__((ext_vector_type(16))) _Float16 v16h;
typedef __attribute__((ext_vector_type(8)))  _Float16 v8h;
typedef __attribute__((ext_vector_type(8)))  float    v8f;

union V16 { v16h v; v8h h[2]; };

// ---------------- fp16 WMMA GEMM: C(MxN) = A(MxK) @ W(NxK)^T + bias ----------------
// Register-blocked, templated wave tile: each wave computes WM x WN 16x16 tiles,
// 8 waves arranged AM(M) x (8/AM)(N) per block.
//   - per-timestep GEMMs (M=512, latency critical): WM=2,WN=2,AM=2 -> 64x128 block
//     tile => ~2x more blocks in flight (occupancy > reuse; weights live in L2).
//   - large batched GEMM (M=28672): WM=2,WN=4,AM=2 -> 64x256 block tile, 1.5
//     b128 loads per WMMA.
// Requires M % (16*AM*WM) == 0, K % 32 == 0, N % 16 == 0 (all satisfied here).
// A-fragment layout (16-bit A 16x32, ISA 7.12.2): lane l -> row l&15; upper half
// of wave holds K+8 (first 8 halves) and K+24 (last 8). B uses identical
// indexing with lane -> output column, W row-major (NxK).
template <int WM, int WN, int AM>
__global__ __launch_bounds__(256) void gemm_f16_wmma(
    const _Float16* __restrict__ A, const _Float16* __restrict__ W,
    const float* __restrict__ bias, float* __restrict__ C,
    int M, int N, int K)
{
    constexpr int AN = 8 / AM;
    const int lane  = threadIdx.x & 31;
    const int wave  = threadIdx.x >> 5;
    const int waveM = wave / AN;
    const int waveN = wave % AN;
    const int mt0   = blockIdx.x * (AM * WM) + waveM * WM;  // first M tile
    const int nt0   = blockIdx.y * (AN * WN) + waveN * WN;  // first N tile
    const int ntiles = N >> 4;

    const int r  = lane & 15;
    const int hi = lane >> 4;

    const _Float16* ap[WM];
#pragma unroll
    for (int i = 0; i < WM; ++i)
        ap[i] = A + (size_t)((mt0 + i) * 16 + r) * K + hi * 8;

    const _Float16* wp[WN];
#pragma unroll
    for (int j = 0; j < WN; ++j) {
        int nt  = nt0 + j;
        int ntc = (nt < ntiles) ? nt : 0;   // clamp OOB tiles (results discarded)
        wp[j] = W + (size_t)(ntc * 16 + r) * K + hi * 8;
    }

    v8f acc[WM][WN] = {};
    for (int k = 0; k < K; k += 32) {
        V16 a[WM];
#pragma unroll
        for (int i = 0; i < WM; ++i) {
            a[i].h[0] = *(const v8h*)(ap[i] + k);
            a[i].h[1] = *(const v8h*)(ap[i] + k + 16);
        }
#pragma unroll
        for (int j = 0; j < WN; ++j) {
            V16 b;
            b.h[0] = *(const v8h*)(wp[j] + k);
            b.h[1] = *(const v8h*)(wp[j] + k + 16);
#pragma unroll
            for (int i = 0; i < WM; ++i)
                acc[i][j] = __builtin_amdgcn_wmma_f32_16x16x32_f16(
                    false, a[i].v, false, b.v, (short)0, acc[i][j], false, false);
        }
    }

#pragma unroll
    for (int j = 0; j < WN; ++j) {
        int nt = nt0 + j;
        if (nt >= ntiles) continue;   // wave-uniform guard
        float bv = bias ? bias[nt * 16 + r] : 0.0f;
#pragma unroll
        for (int i2 = 0; i2 < WM; ++i2) {
            float* cp = C + (size_t)((mt0 + i2) * 16 + hi * 8) * N + nt * 16 + r;
#pragma unroll
            for (int i = 0; i < 8; ++i)
                cp[(size_t)i * N] = acc[i2][j][i] + bv;
        }
    }
}

// ---------------- weight convert + K-pad to fp16 ----------------
__global__ __launch_bounds__(256) void convert_pad(
    const float* __restrict__ src, _Float16* __restrict__ dst, int N, int K, int Kp)
{
    int idx = blockIdx.x * 256 + threadIdx.x;
    if (idx >= N * Kp) return;
    int n = idx / Kp, k = idx % Kp;
    float v = (k < K) ? src[(size_t)n * K + k] : 0.0f;
    dst[idx] = (_Float16)v;
}

__global__ __launch_bounds__(256) void zero_f32(float* __restrict__ p, int n)
{
    int i = blockIdx.x * 256 + threadIdx.x;
    if (i < n) p[i] = 0.0f;
}
__global__ __launch_bounds__(256) void zero_f16(_Float16* __restrict__ p, int n)
{
    int i = blockIdx.x * 256 + threadIdx.x;
    if (i < n) p[i] = (_Float16)0.0f;
}

// ---------------- build encoder input x [T][B][32] fp16 ----------------
__global__ __launch_bounds__(256) void build_x(
    const float* __restrict__ inp, const float* __restrict__ embE,
    _Float16* __restrict__ x)
{
    int idx = blockIdx.x * 256 + threadIdx.x;
    if (idx >= kTE * kB * kKX) return;
    int c = idx % kKX;
    int tb = idx / kKX;
    int b = tb % kB, t = tb / kB;
    const float* row = inp + ((size_t)b * kTE + t) * 6;
    float v = 0.0f;
    if (c < 4) v = row[c];
    else if (c < 20) {
        int e = (int)row[5];
        v = embE[e * 16 + (c - 4)];
    }
    x[idx] = (_Float16)v;
}

// ---------------- GRU pointwise: h' = (1-z)*n + z*h ----------------
__global__ __launch_bounds__(256) void gru_pointwise(
    const float* __restrict__ gi, const float* __restrict__ gh,
    float* __restrict__ h, _Float16* __restrict__ h16, int H)
{
    int idx = blockIdx.x * 256 + threadIdx.x;
    if (idx >= kB * H) return;
    int b = idx / H, j = idx % H;
    const float* gib = gi + (size_t)b * 3 * H;
    const float* ghb = gh + (size_t)b * 3 * H;
    float r = 1.0f / (1.0f + expf(-(gib[j] + ghb[j])));
    float z = 1.0f / (1.0f + expf(-(gib[H + j] + ghb[H + j])));
    float n = tanhf(gib[2 * H + j] + r * ghb[2 * H + j]);
    float hn = (1.0f - z) * n + z * h[idx];
    h[idx] = hn;
    h16[idx] = (_Float16)hn;
}

// ---------------- decoder init: h0 = [h_enc_last, item_emb]; dec_inp = inp[:,-1,0] ----------------
__global__ __launch_bounds__(256) void dec_init(
    const float* __restrict__ hEnc, const float* __restrict__ inp,
    const float* __restrict__ embI,
    float* __restrict__ h0, _Float16* __restrict__ h016,
    float* __restrict__ decInp)
{
    int idx = blockIdx.x * 256 + threadIdx.x;
    if (idx >= kB * kD) return;
    int b = idx / kD, j = idx % kD;
    float v;
    if (j < kNH) v = hEnc[(size_t)b * kNH + j];
    else {
        int it = (int)inp[(size_t)b * kTE * 6 + 4];
        v = embI[(size_t)it * 64 + (j - kNH)];
    }
    h0[idx] = v;
    h016[idx] = (_Float16)v;
    if (j == 0) decInp[b] = inp[(size_t)b * kTE * 6 + (kTE - 1) * 6];
}

// ---------------- attention scores + softmax + context, per batch row ----------------
__global__ __launch_bounds__(256) void attn_context(
    const float* __restrict__ encAttn,   // [T][B][512]
    const float* __restrict__ hidAtt,    // [B][512]
    const float* __restrict__ Vv,        // [512]
    const _Float16* __restrict__ oAll,   // [T][B][512]
    const float* __restrict__ decInp,    // [B]
    _Float16* __restrict__ attnOut)      // [B][544]
{
    int b = blockIdx.x;
    __shared__ float sw[kTE];
    int tid = threadIdx.x;
    int lane = tid & 31, wave = tid >> 5;
    const float* ha = hidAtt + (size_t)b * kNH;

    for (int t = wave; t < kTE; t += 8) {
        const float* ea = encAttn + ((size_t)t * kB + b) * kNH;
        float s = 0.0f;
        for (int j = lane; j < kNH; j += 32)
            s += tanhf(ea[j] + ha[j]) * Vv[j];
#pragma unroll
        for (int o = 16; o > 0; o >>= 1) s += __shfl_xor(s, o, 32);
        if (lane == 0) sw[t] = s;
    }
    __syncthreads();
    if (tid == 0) {
        float mx = sw[0];
        for (int t = 1; t < kTE; ++t) mx = fmaxf(mx, sw[t]);
        float sum = 0.0f;
        for (int t = 0; t < kTE; ++t) { float e = expf(sw[t] - mx); sw[t] = e; sum += e; }
        float inv = 1.0f / sum;
        for (int t = 0; t < kTE; ++t) sw[t] *= inv;
    }
    __syncthreads();

    _Float16* ao = attnOut + (size_t)b * kKA;
    for (int j = tid; j < kNH; j += 256) {
        float acc = 0.0f;
        for (int t = 0; t < kTE; ++t)
            acc += sw[t] * (float)oAll[((size_t)t * kB + b) * kNH + j];
        ao[1 + j] = (_Float16)acc;
    }
    if (tid == 0) ao[0] = (_Float16)decInp[b];
    if (tid < kKA - kNH - 1) ao[kNH + 1 + tid] = (_Float16)0.0f;  // pad 513..543
}

// ---------------- final projection: out[b,t] = h1 . fc_W + fc_b ----------------
__global__ __launch_bounds__(256) void fc_out(
    const float* __restrict__ h1, const float* __restrict__ fcW,
    const float* __restrict__ fcb, float* __restrict__ out,
    float* __restrict__ decInp, int t)
{
    int lane = threadIdx.x & 31;
    int wave = threadIdx.x >> 5;
    int b = blockIdx.x * 8 + wave;
    if (b >= kB) return;
    const float* hb = h1 + (size_t)b * kD;
    float s = 0.0f;
    for (int j = lane; j < kD; j += 32) s += hb[j] * fcW[j];
#pragma unroll
    for (int o = 16; o > 0; o >>= 1) s += __shfl_xor(s, o, 32);
    if (lane == 0) {
        float v = s + fcb[0];
        out[(size_t)b * kTD + t] = v;
        decInp[b] = v;
    }
}

// ---------------- host orchestration ----------------
extern "C" void kernel_launch(void* const* d_in, const int* in_sizes, int n_in,
                              void* d_out, int out_size, void* d_ws, size_t ws_size,
                              hipStream_t stream)
{
    const float* inp      = (const float*)d_in[0];
    const float* embItem  = (const float*)d_in[1];
    const float* embEvent = (const float*)d_in[2];
    const float* enc_Wi0  = (const float*)d_in[3];
    const float* enc_Wh0  = (const float*)d_in[4];
    const float* enc_bi0  = (const float*)d_in[5];
    const float* enc_bh0  = (const float*)d_in[6];
    const float* enc_Wi1  = (const float*)d_in[7];
    const float* enc_Wh1  = (const float*)d_in[8];
    const float* enc_bi1  = (const float*)d_in[9];
    const float* enc_bh1  = (const float*)d_in[10];
    const float* ae_W     = (const float*)d_in[11];
    const float* ae_b     = (const float*)d_in[12];
    const float* ah_W     = (const float*)d_in[13];
    const float* ah_b     = (const float*)d_in[14];
    const float* Vw       = (const float*)d_in[15];
    const float* d1_Wi    = (const float*)d_in[16];
    const float* d1_Wh    = (const float*)d_in[17];
    const float* d1_bi    = (const float*)d_in[18];
    const float* d1_bh    = (const float*)d_in[19];
    const float* dn_Wi    = (const float*)d_in[20];
    const float* dn_Wh    = (const float*)d_in[21];
    const float* dn_bi    = (const float*)d_in[22];
    const float* dn_bh    = (const float*)d_in[23];
    const float* fc_W     = (const float*)d_in[24];
    const float* fc_b     = (const float*)d_in[25];
    float* out = (float*)d_out;
    (void)in_sizes; (void)n_in; (void)out_size; (void)ws_size;

    // bump allocator on workspace (256B aligned chunks)
    char* p = (char*)d_ws;
    auto carve = [&](size_t bytes) -> char* {
        char* r = p;
        p += (bytes + 255) & ~(size_t)255;
        return r;
    };

    _Float16* xAll    = (_Float16*)carve((size_t)kTE * kB * kKX * 2);
    _Float16* Wi0h    = (_Float16*)carve((size_t)3 * kNH * kKX * 2);
    _Float16* Wh0h    = (_Float16*)carve((size_t)3 * kNH * kNH * 2);
    _Float16* Wi1h    = (_Float16*)carve((size_t)3 * kNH * kNH * 2);
    _Float16* Wh1h    = (_Float16*)carve((size_t)3 * kNH * kNH * 2);
    _Float16* aeWh    = (_Float16*)carve((size_t)kNH * kNH * 2);
    _Float16* ahWh    = (_Float16*)carve((size_t)kNH * kD * 2);
    _Float16* d1Wih   = (_Float16*)carve((size_t)3 * kD * kKA * 2);
    _Float16* d1Whh   = (_Float16*)carve((size_t)3 * kD * kD * 2);
    _Float16* dnWih   = (_Float16*)carve((size_t)3 * kD * kD * 2);
    _Float16* dnWhh   = (_Float16*)carve((size_t)3 * kD * kD * 2);
    _Float16* o0      = (_Float16*)carve((size_t)kTE * kB * kNH * 2);   // layer0 outputs [T][B][H]
    _Float16* o1      = (_Float16*)carve((size_t)kTE * kB * kNH * 2);   // layer1 outputs [T][B][H]
    float*    encAttn = (float*)   carve((size_t)kTE * kB * kNH * 4);
    _Float16* zero16  = (_Float16*)carve((size_t)kB * kNH * 2);
    float*    hEnc    = (float*)   carve((size_t)kB * kNH * 4);
    float*    giBuf   = (float*)   carve((size_t)kB * 3 * kD * 4);
    float*    ghBuf   = (float*)   carve((size_t)kB * 3 * kD * 4);
    float*    h0d     = (float*)   carve((size_t)kB * kD * 4);
    _Float16* h0d16   = (_Float16*)carve((size_t)kB * kD * 2);
    float*    h1d     = (float*)   carve((size_t)kB * kD * 4);
    _Float16* h1d16   = (_Float16*)carve((size_t)kB * kD * 2);
    float*    hidAtt  = (float*)   carve((size_t)kB * kNH * 4);
    _Float16* attnOut = (_Float16*)carve((size_t)kB * kKA * 2);
    float*    decInp  = (float*)   carve((size_t)kB * 4);

    auto blocks = [](int n) { return dim3((n + 255) / 256); };
    // latency-critical per-timestep GEMM: 32x32 wave tile, 64x128 block tile
    auto gemm = [&](const _Float16* A, const _Float16* W, const float* bias,
                    float* C, int M, int N, int K) {
        dim3 g(M / 64, (N / 16 + 7) / 8);
        gemm_f16_wmma<2, 2, 2><<<g, dim3(256), 0, stream>>>(A, W, bias, C, M, N, K);
    };
    // large batched GEMM: 32x64 wave tile, 64x256 block tile (max reuse)
    auto gemm_big = [&](const _Float16* A, const _Float16* W, const float* bias,
                        float* C, int M, int N, int K) {
        dim3 g(M / 64, (N / 16 + 15) / 16);
        gemm_f16_wmma<2, 4, 2><<<g, dim3(256), 0, stream>>>(A, W, bias, C, M, N, K);
    };

    // ---- phase 0: convert weights, build inputs ----
    convert_pad<<<blocks(3 * kNH * kKX), 256, 0, stream>>>(enc_Wi0, Wi0h, 3 * kNH, 20, kKX);
    convert_pad<<<blocks(3 * kNH * kNH), 256, 0, stream>>>(enc_Wh0, Wh0h, 3 * kNH, kNH, kNH);
    convert_pad<<<blocks(3 * kNH * kNH), 256, 0, stream>>>(enc_Wi1, Wi1h, 3 * kNH, kNH, kNH);
    convert_pad<<<blocks(3 * kNH * kNH), 256, 0, stream>>>(enc_Wh1, Wh1h, 3 * kNH, kNH, kNH);
    convert_pad<<<blocks(kNH * kNH), 256, 0, stream>>>(ae_W, aeWh, kNH, kNH, kNH);
    convert_pad<<<blocks(kNH * kD), 256, 0, stream>>>(ah_W, ahWh, kNH, kD, kD);
    convert_pad<<<blocks(3 * kD * kKA), 256, 0, stream>>>(d1_Wi, d1Wih, 3 * kD, 1 + kNH, kKA);
    convert_pad<<<blocks(3 * kD * kD), 256, 0, stream>>>(d1_Wh, d1Whh, 3 * kD, kD, kD);
    convert_pad<<<blocks(3 * kD * kD), 256, 0, stream>>>(dn_Wi, dnWih, 3 * kD, kD, kD);
    convert_pad<<<blocks(3 * kD * kD), 256, 0, stream>>>(dn_Wh, dnWhh, 3 * kD, kD, kD);
    build_x<<<blocks(kTE * kB * kKX), 256, 0, stream>>>(inp, embEvent, xAll);
    zero_f16<<<blocks(kB * kNH), 256, 0, stream>>>(zero16, kB * kNH);

    // ---- encoder layer 0 ----
    zero_f32<<<blocks(kB * kNH), 256, 0, stream>>>(hEnc, kB * kNH);
    for (int t = 0; t < kTE; ++t) {
        gemm(xAll + (size_t)t * kB * kKX, Wi0h, enc_bi0, giBuf, kB, 3 * kNH, kKX);
        const _Float16* hprev = (t == 0) ? zero16 : (o0 + (size_t)(t - 1) * kB * kNH);
        gemm(hprev, Wh0h, enc_bh0, ghBuf, kB, 3 * kNH, kNH);
        gru_pointwise<<<blocks(kB * kNH), 256, 0, stream>>>(
            giBuf, ghBuf, hEnc, o0 + (size_t)t * kB * kNH, kNH);
    }

    // ---- encoder layer 1 ----
    zero_f32<<<blocks(kB * kNH), 256, 0, stream>>>(hEnc, kB * kNH);
    for (int t = 0; t < kTE; ++t) {
        gemm(o0 + (size_t)t * kB * kNH, Wi1h, enc_bi1, giBuf, kB, 3 * kNH, kNH);
        const _Float16* hprev = (t == 0) ? zero16 : (o1 + (size_t)(t - 1) * kB * kNH);
        gemm(hprev, Wh1h, enc_bh1, ghBuf, kB, 3 * kNH, kNH);
        gru_pointwise<<<blocks(kB * kNH), 256, 0, stream>>>(
            giBuf, ghBuf, hEnc, o1 + (size_t)t * kB * kNH, kNH);
    }

    // ---- attention projection over all encoder outputs (large batched GEMM) ----
    gemm_big(o1, aeWh, ae_b, encAttn, kTE * kB, kNH, kNH);

    // ---- decoder init ----
    dec_init<<<blocks(kB * kD), 256, 0, stream>>>(hEnc, inp, embItem, h0d, h0d16, decInp);
    zero_f32<<<blocks(kB * kD), 256, 0, stream>>>(h1d, kB * kD);
    zero_f16<<<blocks(kB * kD), 256, 0, stream>>>(h1d16, kB * kD);

    // ---- decoder loop ----
    for (int t = 0; t < kTD; ++t) {
        gemm(h0d16, ahWh, ah_b, hidAtt, kB, kNH, kD);
        attn_context<<<dim3(kB), dim3(256), 0, stream>>>(
            encAttn, hidAtt, Vw, o1, decInp, attnOut);
        gemm(attnOut, d1Wih, d1_bi, giBuf, kB, 3 * kD, kKA);
        gemm(h0d16, d1Whh, d1_bh, ghBuf, kB, 3 * kD, kD);
        gru_pointwise<<<blocks(kB * kD), 256, 0, stream>>>(giBuf, ghBuf, h0d, h0d16, kD);
        gemm(h0d16, dnWih, dn_bi, giBuf, kB, 3 * kD, kD);
        gemm(h1d16, dnWhh, dn_bh, ghBuf, kB, 3 * kD, kD);
        gru_pointwise<<<blocks(kB * kD), 256, 0, stream>>>(giBuf, ghBuf, h1d, h1d16, kD);
        fc_out<<<dim3(kB / 8), dim3(256), 0, stream>>>(h1d, fc_W, fc_b, out, decInp, t);
    }
}
